// FMSweepBankNetwork_54485955117347
// MI455X (gfx1250) — matrix operand order repaired and checked by reading, not compiled, not run on hardware
//
#include <hip/hip_runtime.h>

typedef __attribute__((ext_vector_type(16))) _Float16 v16h;
typedef __attribute__((ext_vector_type(8)))  float    v8f;

#define TT 2048
#define BB 64
#define FF 64
#define NN 64

#define BETA   0.6f
#define TH_V   0.6f
#define REFR_V 2.0f
#define INH_S  0.3f
#define INH_D  0.9f

// ---------------------------------------------------------------------------
// Phase 1: coincidence current via f16 WMMA, diagonal-tile trick.
// One wave (32 threads) handles one (t-chunk, batch b, 16-wide n-tile).
// A[n_local, f] = pulse[t-n, b, f] * echo[t, b, f]   (16x64, f16)
// B[f, n_local] = freq_w[n, f]                        (64x16, f16, t-invariant)
// cur[t,b,n] = diag(A @ B) extracted from the f32 accumulator layout.
// ---------------------------------------------------------------------------
__global__ __launch_bounds__(32) void coincidence_wmma_kernel(
    const float* __restrict__ pulse, const float* __restrict__ echo,
    const float* __restrict__ fw, float* __restrict__ cur)
{
  const int lane = threadIdx.x;   // 0..31
  const int row  = lane & 15;     // A-row / B-col within tile
  const int hi   = lane >> 4;     // hi-half lane group

  const int gid   = blockIdx.x;
  const int ntile = gid & 3;          // 4 n-tiles of 16
  const int b     = (gid >> 2) & 63;  // batch
  const int tc    = gid >> 8;         // t-chunk (16 chunks of 128)
  const int n0    = ntile << 4;

  // ---- pack B operand (freq_w), invariant over t -------------------------
  // B 32x16 as packed here (contiguous k, same as A): lanes 0-15 hold K=0..15 at col N=lane,
  // lanes 16-31 hold K=16..31 at col N=lane-16; halves j map to successive K.
  v16h Bm0, Bm1;
  {
    const float* w = fw + (size_t)(n0 + row) * FF + hi * 16;
    #pragma unroll
    for (int j = 0; j < 16; ++j) {
      Bm0[j] = (_Float16)w[j];        // f chunk 0: f = hi*16 + j
      Bm1[j] = (_Float16)w[32 + j];   // f chunk 1: f = 32 + hi*16 + j
    }
  }

  const int n  = n0 + row;            // this lane's A-row == its delay
  const int t0 = tc * (TT / 16);
  const int t1 = t0 + (TT / 16);

  for (int t = t0; t < t1; ++t) {
    const float* e = echo + ((size_t)t * BB + b) * FF;

    // ---- pack A operand: 16-bit A 16x32 layout ---------------------------
    // lane (row, hi): halves j=0..15 -> K = hi*16 + j (matches B packing)
    v16h A0 = {}, A1 = {};
    const int ts = t - n;             // delayed time index; <0 -> zeros
    if (ts >= 0) {
      const float* p = pulse + ((size_t)ts * BB + b) * FF;
      #pragma unroll
      for (int j = 0; j < 16; ++j) {
        const int K = hi * 16 + j;
        A0[j] = (_Float16)(p[K] * e[K]);
        A1[j] = (_Float16)(p[32 + K] * e[32 + K]);
      }
    }

    // ---- D = A*B + C, K=64 via two chained 16x16x32 f16 WMMAs ------------
    v8f c = {};
    c = __builtin_amdgcn_wmma_f32_16x16x32_f16(false, A0, false, Bm0,
                                               (short)0, c, false, false);
    c = __builtin_amdgcn_wmma_f32_16x16x32_f16(false, A1, false, Bm1,
                                               (short)0, c, false, false);

    // ---- extract diagonal C[n_local, n_local] ----------------------------
    // C layout: VGPR r, lane l -> (M = r + 8*(l>=16), N = l&15).
    // Diag element n_local lives in lane l where (l&15)==n_local and
    // (l>=16) == (n_local>=8), at VGPR index n_local&7.
    if (hi == (row >> 3)) {
      cur[((size_t)t * BB + b) * NN + n] = c[row & 7];
    }

    if (t + 1 < t1) __builtin_prefetch(e + BB * FF, 0, 1);  // next echo row
  }
}

// ---------------------------------------------------------------------------
// Phase 2: sequential LIF scan. One workgroup per batch, one thread per
// neuron n; spikes exchanged through LDS for the radius-2 lateral drive.
// ---------------------------------------------------------------------------
__global__ __launch_bounds__(NN) void lif_scan_kernel(
    const float* __restrict__ cur, float* __restrict__ spk_out)
{
  const int b = blockIdx.x;
  const int n = threadIdx.x;

  __shared__ float spk_sh[NN];

  float mem = 0.f, refr = 0.f, inh = 0.f;

  for (int t = 0; t < TT; ++t) {
    const float c = cur[((size_t)t * BB + b) * NN + n];

    const float reset      = (mem > TH_V) ? 1.f : 0.f;
    const float mem_upd    = BETA * mem + c - reset * TH_V;
    const float spk_active = (mem_upd > TH_V) ? 1.f : 0.f;
    const bool  active     = (refr <= 0.f);
    const float spk        = active ? spk_active : 0.f;
    float       mem_new    = active ? mem_upd : BETA * mem;

    refr = (spk > 0.f) ? REFR_V : fmaxf(refr - 1.f, 0.f);

    spk_sh[n] = spk;
    __syncthreads();

    float drive = 0.f;
    #pragma unroll
    for (int o = 1; o <= 2; ++o) {
      const float w = 1.f / (float)o;
      if (n - o >= 0) drive += w * spk_sh[n - o];
      if (n + o < NN) drive += w * spk_sh[n + o];
    }
    __syncthreads();   // protect spk_sh before next iteration's write

    inh = INH_D * inh + drive;
    mem_new -= INH_S * inh;
    mem = mem_new;

    spk_out[((size_t)t * BB + b) * NN + n] = spk;
  }
}

// ---------------------------------------------------------------------------
extern "C" void kernel_launch(void* const* d_in, const int* in_sizes, int n_in,
                              void* d_out, int out_size, void* d_ws, size_t ws_size,
                              hipStream_t stream) {
  const float* pulse = (const float*)d_in[0];  // [T,B,F]
  const float* echo  = (const float*)d_in[1];  // [T,B,F]
  const float* fw    = (const float*)d_in[2];  // [N,F]

  float* cur = (float*)d_ws;                   // [T,B,N] scratch (32 MB)
  float* out = (float*)d_out;                  // [T,B,N] spikes

  // 16 t-chunks x 64 batches x 4 n-tiles = 4096 single-wave blocks
  coincidence_wmma_kernel<<<16 * 64 * 4, 32, 0, stream>>>(pulse, echo, fw, cur);

  // 64 batches x 64 neurons, sequential over T (same stream -> ordered)
  lif_scan_kernel<<<BB, NN, 0, stream>>>(cur, out);
}
